// MediumRangeEdge_11072425689094
// MI455X (gfx1250) — matrix operand order repaired and checked by reading, not compiled, not run on hardware
//
#include <hip/hip_runtime.h>
#include <hip/hip_bf16.h>

#define DIMK   512
#define RESO   28
#define NODES  784          // RESO*RESO
#define KSEL   10
#define BATCH  32
#define INF_MASK 100000.0f
#define DSTRIDE 788         // 784 padded; 8*788 % 64 != 0 -> fewer LDS bank conflicts

typedef __attribute__((ext_vector_type(16))) _Float16 v16h;
typedef __attribute__((ext_vector_type(8)))  float    v8f;
typedef __attribute__((ext_vector_type(4)))  float    v4f;

union HalfVec {
    v16h h;
    v4f  f[2];
};

// ---------------------------------------------------------------------------
// Kernel 1: L2-normalize each 512-float row, store as fp16 (one wave per row).
// ---------------------------------------------------------------------------
__global__ __launch_bounds__(256)
void normalize_f16_kernel(const float* __restrict__ x, _Float16* __restrict__ xh)
{
    const int wavesPerBlk = blockDim.x >> 5;
    const int row  = blockIdx.x * wavesPerBlk + (threadIdx.x >> 5);
    const int lane = threadIdx.x & 31;

    const float* xr = x + (size_t)row * DIMK;
    float v[16];
    float ss = 0.f;
#pragma unroll
    for (int i = 0; i < 16; ++i) {
        v[i] = xr[lane + 32 * i];
        ss += v[i] * v[i];
    }
#pragma unroll
    for (int off = 16; off > 0; off >>= 1)
        ss += __shfl_xor(ss, off, 32);

    const float inv = 1.0f / fmaxf(sqrtf(ss), 1e-12f);

    _Float16* xo = xh + (size_t)row * DIMK;
#pragma unroll
    for (int i = 0; i < 16; ++i)
        xo[lane + 32 * i] = (_Float16)(v[i] * inv);
}

// ---------------------------------------------------------------------------
// Kernel 2: per (batch, 16-row strip): Gram tile via WMMA, bias+mask epilogue
// into an LDS distance strip, then K=10 wave-wide argmin selection.
// grid = (49, BATCH), block = 256 (8 waves).
// ---------------------------------------------------------------------------
__global__ __launch_bounds__(256)
void gram_topk_kernel(const _Float16* __restrict__ xh,
                      const float*    __restrict__ rel,
                      int*            __restrict__ out)
{
    __shared__ float dist[16 * DSTRIDE];   // ~50 KB

    const int b    = blockIdx.y;
    const int rt   = blockIdx.x;           // 0..48 row tile
    const int row0 = rt * 16;
    const int tid  = threadIdx.x;
    const int wave = tid >> 5;
    const int lane = tid & 31;
    const int mn   = lane & 15;            // A row / B column within 16x16 tile
    const int h    = lane >> 4;            // lane half (ISA fragment layout)

    const _Float16* Xb    = xh + (size_t)b * NODES * DIMK;
    // A fragment base: row (row0+mn), K chunks at h*8 and 16+h*8 (per ISA 16-bit A layout)
    const _Float16* Abase = Xb + (size_t)(row0 + mn) * DIMK + h * 8;

    // ---- Gram + epilogue: each wave walks column tiles ct = wave, wave+8, ...
    for (int ct = wave; ct < 49; ct += 8) {
        const int col0 = ct * 16;
        // B fragment base: column-node row (col0+mn), K chunks at h*8 and 16+h*8
        const _Float16* Bbase = Xb + (size_t)(col0 + mn) * DIMK + h * 8;

        v8f acc = {0.f, 0.f, 0.f, 0.f, 0.f, 0.f, 0.f, 0.f};
#pragma unroll 4
        for (int ks = 0; ks < 16; ++ks) {          // K = 512 = 16 * 32
            const int k0 = ks * 32;
            HalfVec a, bb;
            a.f[0]  = *(const v4f*)(Abase + k0);        // K = k0 + h*8 .. +8
            a.f[1]  = *(const v4f*)(Abase + k0 + 16);   // K = k0+16+h*8 .. +8
            bb.f[0] = *(const v4f*)(Bbase + k0);        // K = k0 + h*8 .. +8
            bb.f[1] = *(const v4f*)(Bbase + k0 + 16);   // K = k0+16+h*8 .. +8
            acc = __builtin_amdgcn_wmma_f32_16x16x32_f16(
                false, a.h, false, bb.h, (short)0, acc, false, false);
        }

        // Epilogue: dist = 2 - 2*dot + rel + INF*neighbor_mask
        const int jg = col0 + mn;              // column node index
        const int jr = jg / RESO, jc = jg % RESO;
#pragma unroll
        for (int i = 0; i < 8; ++i) {
            const int ml = i + 8 * h;          // C/D layout: M = vgpr + 8*half
            const int ig = row0 + ml;          // row node index
            float d = 2.0f - 2.0f * acc[i] + rel[ig * NODES + jg];
            const int dr = (ig / RESO) - jr;
            const int dc = (ig % RESO) - jc;
            if (dr >= -1 && dr <= 1 && dc >= -1 && dc <= 1)
                d += INF_MASK;                 // diagonal + 8 spatial neighbors
            dist[ml * DSTRIDE + jg] = d;
        }
    }

    __syncthreads();

    // ---- Top-K selection: each wave owns rows wave and wave+8.
    for (int rr = wave; rr < 16; rr += 8) {
        const int ig   = row0 + rr;
        float*    drow = &dist[rr * DSTRIDE];
        for (int k = 0; k < KSEL; ++k) {
            float best = 3.4e38f;
            int   bidx = NODES;
            for (int j = lane; j < NODES; j += 32) {
                const float v = drow[j];
                if (v < best) { best = v; bidx = j; }   // keeps lowest j on ties
            }
#pragma unroll
            for (int off = 16; off > 0; off >>= 1) {
                const float ov = __shfl_xor(best, off, 32);
                const int   oi = __shfl_xor(bidx, off, 32);
                if (ov < best || (ov == best && oi < bidx)) { best = ov; bidx = oi; }
            }
            if (lane == 0) {
                drow[bidx] = 3.0e38f;          // exclude from later rounds
                const int e = ((b * NODES + ig) * KSEL + k) * 3;
                out[e + 0] = b * NODES + bidx; // dst
                out[e + 1] = b * NODES + ig;   // src
                out[e + 2] = 0;                // relation
            }
            // make lane-0's LDS marker visible to the whole wave's next sweep
            asm volatile("s_wait_dscnt 0" ::: "memory");
        }
    }
}

extern "C" void kernel_launch(void* const* d_in, const int* in_sizes, int n_in,
                              void* d_out, int out_size, void* d_ws, size_t ws_size,
                              hipStream_t stream)
{
    const float* node = (const float*)d_in[0];   // [B, N, D] fp32
    const float* rel  = (const float*)d_in[1];   // [1, N, N] fp32
    int*         out  = (int*)d_out;             // [B*N*K, 3] int32
    _Float16*    xh   = (_Float16*)d_ws;         // normalized fp16, B*N*D (25.7 MB)

    const int rows = BATCH * NODES;              // 25088 rows, 8 waves/block
    normalize_f16_kernel<<<rows / 8, 256, 0, stream>>>(node, xh);

    dim3 grid(49, BATCH);
    gram_topk_kernel<<<grid, 256, 0, stream>>>(xh, rel, out);
}
